// MultiHeadSelfAttention_77532749627666
// MI455X (gfx1250) — compile-verified, matrix operand order checked by reading, not run on hardware
//
#include <hip/hip_runtime.h>
#include <hip/hip_bf16.h>

typedef _Float16 half_t;
typedef _Float16 v16h __attribute__((ext_vector_type(16)));
typedef _Float16 v8h  __attribute__((ext_vector_type(8)));
typedef float    v8f  __attribute__((ext_vector_type(8)));
typedef unsigned int v4u __attribute__((ext_vector_type(4)));
typedef int          v4i __attribute__((ext_vector_type(4)));
typedef int          v8i __attribute__((ext_vector_type(8)));

#define B_NUM 4
#define T_SEQ 2048
#define C_DIM 1024
#define H_NUM 16
#define D_DIM 64

// ---------------------------------------------------------------------------
// WMMA helper: D = A(16x32 f16) * B(32x16 f16) + C(16x16 f32)
// ---------------------------------------------------------------------------
__device__ __forceinline__ v8f wmma32(v16h a, v16h b, v8f c) {
    return __builtin_amdgcn_wmma_f32_16x16x32_f16(false, a, false, b, (short)0, c,
                                                  false, false);
}

// Load one 16x32 fragment from global memory (A layout; B is the mirrored
// column read, identical access pattern for row-major-along-K operands).
__device__ __forceinline__ v16h load_frag(const half_t* __restrict__ p, int ld,
                                          int row0, int k0) {
    const int lane = threadIdx.x & 31;
    const int r    = row0 + (lane & 15);
    const int kb   = k0 + ((lane >> 4) << 3);
    const half_t* base = p + (size_t)r * (size_t)ld + kb;
    union { v16h v; v8h h[2]; } f;
    f.h[0] = *(const v8h*)(base);
    f.h[1] = *(const v8h*)(base + 16);
    return f.v;
}

// Fragment load from a __shared__ 2-buffer array, indexed DIRECTLY so clang's
// InferAddressSpaces proves addrspace(3) and emits ds_load_b128 (not flat).
#define LOAD_FRAG_LDS(dstv, arr, bi, row0_, k0_)                               \
    {                                                                          \
        const int lane_ = threadIdx.x & 31;                                    \
        const int r_    = (row0_) + (lane_ & 15);                              \
        const int kb_   = (k0_) + ((lane_ >> 4) << 3);                         \
        union { v16h v; v8h h[2]; } f_;                                        \
        f_.h[0] = *(const v8h*)&(arr)[bi][r_ * 128 + kb_];                     \
        f_.h[1] = *(const v8h*)&(arr)[bi][r_ * 128 + kb_ + 16];                \
        dstv = f_.v;                                                           \
    }

__device__ __forceinline__ float red_max16(float v) {
    v = fmaxf(v, __shfl_xor(v, 1, 32));
    v = fmaxf(v, __shfl_xor(v, 2, 32));
    v = fmaxf(v, __shfl_xor(v, 4, 32));
    v = fmaxf(v, __shfl_xor(v, 8, 32));
    return v;
}
__device__ __forceinline__ float red_sum16(float v) {
    v += __shfl_xor(v, 1, 32);
    v += __shfl_xor(v, 2, 32);
    v += __shfl_xor(v, 4, 32);
    v += __shfl_xor(v, 8, 32);
    return v;
}

// ---------------------------------------------------------------------------
// TDM: stage a 64-row x 128-col f16 tile (row stride C_DIM) into LDS.
// Tracked with TENSORcnt; EXEC ignored (one issue per wave).
// ---------------------------------------------------------------------------
__device__ __forceinline__ void tdm_load_64x128(const half_t* gsrc,
                                                unsigned lds_addr) {
    const unsigned long long ga = (unsigned long long)(uintptr_t)gsrc;
    v4u g0;
    g0[0] = 1u;                                   // count=1 (valid user D#)
    g0[1] = lds_addr;                             // LDS byte address
    g0[2] = (unsigned)ga;                         // global_addr[31:0]
    g0[3] = ((unsigned)(ga >> 32) & 0x01FFFFFFu) | 0x80000000u; // addr hi | type=2
    v8i g1;
    g1[0] = 0x00010000;                           // workgroup_mask=0, data_size=2B
    g1[1] = (int)(((unsigned)C_DIM & 0xFFFFu) << 16);                 // dim0 lo16
    g1[2] = (int)(((unsigned)C_DIM >> 16) | (8192u & 0xFFFFu) << 16); // dim0 hi|dim1 lo
    g1[3] = (int)((8192u >> 16) | (128u << 16));  // dim1 hi | tile_dim0=128
    g1[4] = 64;                                   // tile_dim1=64, tile_dim2=0
    g1[5] = C_DIM;                                // tensor_dim0_stride lo32
    g1[6] = 0;
    g1[7] = 0;
    v4i z4 = {0, 0, 0, 0};
#if __has_include(<hip/amd_detail/amd_gfx1250_TDM.h>)
    v8i z8 = {0, 0, 0, 0, 0, 0, 0, 0};
    __builtin_amdgcn_tensor_load_to_lds(g0, g1, z4, z4, z8, 0);
#else
    __builtin_amdgcn_tensor_load_to_lds(g0, g1, z4, z4, 0);
#endif
}

// ---------------------------------------------------------------------------
// Shared GEMM K-loop: 64x64 block tile, wave tile 64x16. The activation tile
// (shared by all 4 waves) is TDM-staged 64x128 per K-chunk, double buffered;
// the 16 weight rows per wave stream from global. Direct shared-array indexing
// keeps the A path on ds_load_b128.
// ---------------------------------------------------------------------------
#define GEMM_CORE(Aptr, Wptr, row0_, n0_, wave_, arr, accv)                    \
    {                                                                          \
        const unsigned lb_[2] = {(unsigned)(uintptr_t)&(arr)[0][0],            \
                                 (unsigned)(uintptr_t)&(arr)[1][0]};           \
        const half_t* asrc_ = (Aptr) + (size_t)(row0_) * C_DIM;                \
        if ((wave_) == 0) tdm_load_64x128(asrc_, lb_[0]);                      \
        for (int c_ = 0; c_ < C_DIM / 128; ++c_) {                             \
            if ((wave_) == 0) {                                                \
                if (c_ + 1 < C_DIM / 128) {                                    \
                    tdm_load_64x128(asrc_ + (c_ + 1) * 128, lb_[(c_ + 1) & 1]);\
                    __builtin_amdgcn_s_wait_tensorcnt((short)1);               \
                } else {                                                       \
                    __builtin_amdgcn_s_wait_tensorcnt((short)0);               \
                }                                                              \
            }                                                                  \
            __syncthreads();                     /* chunk c_ visible */        \
            const int bi_ = c_ & 1;                                            \
            _Pragma("unroll")                                                  \
            for (int kk_ = 0; kk_ < 128; kk_ += 32) {                          \
                v16h bf_ = load_frag((Wptr), C_DIM, (n0_), c_ * 128 + kk_);    \
                _Pragma("unroll")                                              \
                for (int mi_ = 0; mi_ < 4; ++mi_) {                            \
                    v16h af_;                                                  \
                    LOAD_FRAG_LDS(af_, arr, bi_, mi_ * 16, kk_);               \
                    accv[mi_] = wmma32(af_, bf_, accv[mi_]);                   \
                }                                                              \
            }                                                                  \
            __syncthreads();                     /* reads done before reuse */ \
        }                                                                      \
    }

// ---------------------------------------------------------------------------
// fp32 -> fp16 conversion (8 elements / thread, vectorized)
// ---------------------------------------------------------------------------
__global__ void __launch_bounds__(256) cvt_f32_f16(const float* __restrict__ src,
                                                   half_t* __restrict__ dst, int n) {
    int i = (blockIdx.x * 256 + threadIdx.x) * 8;
    if (i >= n) return;
    float4 a = *(const float4*)(src + i);
    float4 b = *(const float4*)(src + i + 4);
    v8h o;
    o[0] = (half_t)a.x; o[1] = (half_t)a.y; o[2] = (half_t)a.z; o[3] = (half_t)a.w;
    o[4] = (half_t)b.x; o[5] = (half_t)b.y; o[6] = (half_t)b.z; o[7] = (half_t)b.w;
    *(v8h*)(dst + i) = o;
}

// ---------------------------------------------------------------------------
// QKV projection: y = x @ W^T. blockIdx.z: 0 -> K [B,H,T,D], 1 -> Q [B,H,T,D],
// 2 -> V transposed [B,H,D,T] (contiguous B-fragments for PV).
// ---------------------------------------------------------------------------
__global__ void __launch_bounds__(128)
gemm_qkv(const half_t* __restrict__ x16,
         const half_t* __restrict__ wk16, const half_t* __restrict__ wq16,
         const half_t* __restrict__ wv16,
         half_t* __restrict__ k16, half_t* __restrict__ q16,
         half_t* __restrict__ vT16) {
    __shared__ __align__(16) half_t alds[2][64 * 128];

    const int wave = threadIdx.x >> 5;
    const int lane = threadIdx.x & 31;
    const int which = blockIdx.z;
    const half_t* __restrict__ W = (which == 0) ? wk16 : (which == 1) ? wq16 : wv16;

    const int row0 = blockIdx.x * 64;
    const int n0   = blockIdx.y * 64 + wave * 16;

    v8f acc[4] = {};
    GEMM_CORE(x16, W, row0, n0, wave, alds, acc);

    const int nlo = lane & 15;
    const int mhi = (lane >> 4) << 3;
    const int n = n0 + nlo;
    const int h = n >> 6;
    const int d = n & 63;

    if (which == 2) {
        #pragma unroll
        for (int mi = 0; mi < 4; ++mi) {
            const int m0 = row0 + mi * 16 + mhi;
            const int b  = m0 >> 11;
            const int t0 = m0 & (T_SEQ - 1);
            v8h pk;
            #pragma unroll
            for (int r = 0; r < 8; ++r) pk[r] = (half_t)acc[mi][r];
            *(v8h*)(vT16 + (((size_t)(b * H_NUM + h) * D_DIM + d) << 11) + t0) = pk;
        }
    } else {
        half_t* __restrict__ dst = (which == 0) ? k16 : q16;
        #pragma unroll
        for (int mi = 0; mi < 4; ++mi) {
            #pragma unroll
            for (int r = 0; r < 8; ++r) {
                const int m = row0 + mi * 16 + r + mhi;
                const int b = m >> 11;
                const int t = m & (T_SEQ - 1);
                dst[(((size_t)(b * H_NUM + h) << 11) + t) * D_DIM + d] =
                    (half_t)acc[mi][r];
            }
        }
    }
}

// ---------------------------------------------------------------------------
// Flash attention: one wave per (b,h, 16-row q tile). Causal-only S loop in
// blocks of 32 columns; online softmax; P staged through LDS to reshape the
// C-layout score tile into an A-layout 16x32 fragment; PV via transposed V.
// ---------------------------------------------------------------------------
__global__ void __launch_bounds__(128)
attn(const half_t* __restrict__ q16, const half_t* __restrict__ k16,
     const half_t* __restrict__ vT16, half_t* __restrict__ att16) {
    __shared__ __align__(16) half_t lds[4][16 * 32];

    const int wave = threadIdx.x >> 5;
    const int lane = threadIdx.x & 31;
    const int bh   = blockIdx.y;
    const int qtile = blockIdx.x * 4 + wave;
    const int q0 = qtile * 16;

    const size_t bh_off = (size_t)bh * T_SEQ * D_DIM;
    const half_t* __restrict__ Q  = q16 + bh_off;
    const half_t* __restrict__ K  = k16 + bh_off;
    const half_t* __restrict__ VT = vT16 + bh_off;

    const v16h qa0 = load_frag(Q, D_DIM, q0, 0);
    const v16h qa1 = load_frag(Q, D_DIM, q0, 32);

    v8f o[4] = {};
    float mrow[8], srow[8];
    #pragma unroll
    for (int r = 0; r < 8; ++r) { mrow[r] = -INFINITY; srow[r] = 0.f; }

    const int nlo = lane & 15;
    const int mhi = (lane >> 4) << 3;
    const int sblocks = (q0 + 16 + 31) >> 5;

    for (int sb = 0; sb < sblocks; ++sb) {
        const int s0 = sb * 32;

        v8f sv0 = {}, sv1 = {};
        {
            v16h kb0 = load_frag(K, D_DIM, s0, 0);
            v16h kb1 = load_frag(K, D_DIM, s0, 32);
            sv0 = wmma32(qa0, kb0, sv0);
            sv0 = wmma32(qa1, kb1, sv0);
        }
        {
            v16h kb0 = load_frag(K, D_DIM, s0 + 16, 0);
            v16h kb1 = load_frag(K, D_DIM, s0 + 16, 32);
            sv1 = wmma32(qa0, kb0, sv1);
            sv1 = wmma32(qa1, kb1, sv1);
        }

        const int ts0 = s0 + nlo;
        const int ts1 = s0 + 16 + nlo;
        #pragma unroll
        for (int r = 0; r < 8; ++r) {
            const int tq = q0 + r + mhi;
            float a0 = sv0[r] * 0.125f;
            float a1 = sv1[r] * 0.125f;
            a0 = (ts0 > tq) ? -INFINITY : a0;
            a1 = (ts1 > tq) ? -INFINITY : a1;
            const float tmax = red_max16(fmaxf(a0, a1));
            const float nm = fmaxf(mrow[r], tmax);
            const float corr = __expf(mrow[r] - nm);
            mrow[r] = nm;
            const float p0 = __expf(a0 - nm);
            const float p1 = __expf(a1 - nm);
            srow[r] = srow[r] * corr + red_sum16(p0 + p1);
            o[0][r] *= corr; o[1][r] *= corr; o[2][r] *= corr; o[3][r] *= corr;
            lds[wave][(r + mhi) * 32 + nlo]      = (half_t)p0;
            lds[wave][(r + mhi) * 32 + 16 + nlo] = (half_t)p1;
        }

        union { v16h v; v8h h[2]; } pa;
        {
            const int m  = nlo;
            const int kb = mhi;
            pa.h[0] = *(const v8h*)&lds[wave][m * 32 + kb];
            pa.h[1] = *(const v8h*)&lds[wave][m * 32 + 16 + kb];
        }

        #pragma unroll
        for (int c = 0; c < 4; ++c) {
            v16h vb = load_frag(VT, T_SEQ, c * 16, s0);
            o[c] = wmma32(pa.v, vb, o[c]);
        }
    }

    const int b = bh >> 4;
    const int h = bh & 15;
    #pragma unroll
    for (int c = 0; c < 4; ++c) {
        const int d = c * 16 + nlo;
        #pragma unroll
        for (int r = 0; r < 8; ++r) {
            const int t = q0 + r + mhi;
            const float val = o[c][r] / srow[r];
            att16[((size_t)(b * T_SEQ + t)) * C_DIM + h * D_DIM + d] = (half_t)val;
        }
    }
}

// ---------------------------------------------------------------------------
// Output projection: out = att @ Wp^T + bp  (fp32 result)
// ---------------------------------------------------------------------------
__global__ void __launch_bounds__(128)
gemm_proj(const half_t* __restrict__ att16, const half_t* __restrict__ wp16,
          const float* __restrict__ bp, float* __restrict__ out) {
    __shared__ __align__(16) half_t alds[2][64 * 128];

    const int wave = threadIdx.x >> 5;
    const int lane = threadIdx.x & 31;
    const int row0 = blockIdx.x * 64;
    const int n0   = blockIdx.y * 64 + wave * 16;

    v8f acc[4] = {};
    GEMM_CORE(att16, wp16, row0, n0, wave, alds, acc);

    const int nlo = lane & 15;
    const int mhi = (lane >> 4) << 3;
    const int n = n0 + nlo;
    const float bias = bp[n];
    #pragma unroll
    for (int mi = 0; mi < 4; ++mi) {
        #pragma unroll
        for (int r = 0; r < 8; ++r) {
            const int m = row0 + mi * 16 + r + mhi;
            out[(size_t)m * C_DIM + n] = acc[mi][r] + bias;
        }
    }
}

// ---------------------------------------------------------------------------
extern "C" void kernel_launch(void* const* d_in, const int* in_sizes, int n_in,
                              void* d_out, int out_size, void* d_ws, size_t ws_size,
                              hipStream_t stream) {
    const float* x  = (const float*)d_in[0];
    const float* Wk = (const float*)d_in[1];
    const float* Wq = (const float*)d_in[2];
    const float* Wv = (const float*)d_in[3];
    const float* Wp = (const float*)d_in[4];
    const float* bp = (const float*)d_in[5];
    float* out = (float*)d_out;

    const size_t SX = (size_t)B_NUM * T_SEQ * C_DIM;
    const size_t SW = (size_t)C_DIM * C_DIM;

    half_t* ws   = (half_t*)d_ws;
    half_t* x16  = ws;
    half_t* wk16 = x16  + SX;
    half_t* wq16 = wk16 + SW;
    half_t* wv16 = wq16 + SW;
    half_t* wp16 = wv16 + SW;
    half_t* q16  = wp16 + SW;
    half_t* k16  = q16  + SX;
    half_t* vT16 = k16  + SX;
    half_t* a16  = vT16 + SX;

    {
        int nx = (int)SX, nw = (int)SW;
        int bx = (nx / 8 + 255) / 256, bw = (nw / 8 + 255) / 256;
        cvt_f32_f16<<<bx, 256, 0, stream>>>(x,  x16,  nx);
        cvt_f32_f16<<<bw, 256, 0, stream>>>(Wk, wk16, nw);
        cvt_f32_f16<<<bw, 256, 0, stream>>>(Wq, wq16, nw);
        cvt_f32_f16<<<bw, 256, 0, stream>>>(Wv, wv16, nw);
        cvt_f32_f16<<<bw, 256, 0, stream>>>(Wp, wp16, nw);
    }

    gemm_qkv<<<dim3((B_NUM * T_SEQ) / 64, C_DIM / 64, 3), 128, 0, stream>>>(
        x16, wk16, wq16, wv16, k16, q16, vT16);

    attn<<<dim3(T_SEQ / 64, B_NUM * H_NUM), 128, 0, stream>>>(q16, k16, vT16, a16);

    gemm_proj<<<dim3((B_NUM * T_SEQ) / 64, C_DIM / 64), 128, 0, stream>>>(
        a16, wp16, bp, out);
}
